// MultiheadAttention_36782099923659
// MI455X (gfx1250) — compile-verified, matrix operand order checked by reading, not run on hardware
//
#include <hip/hip_runtime.h>
#include <cstdint>

#define B_   4
#define N_   2048
#define E_   1024
#define H_   16
#define HD_  64
#define C3E  3072      // 3*E row stride of qkv_w / qkv_b
#define CHUNK 64       // k rows staged per iteration
#define QT   128       // q rows per block (8 waves * 16)

typedef __attribute__((ext_vector_type(16))) __bf16 v16bf;
typedef __attribute__((ext_vector_type(8)))  float  v8f;

// ---------------------------------------------------------------- TDM helper
#if __has_builtin(__builtin_amdgcn_tensor_load_to_lds)
#define USE_TDM 1
typedef __attribute__((ext_vector_type(4))) unsigned int tdm4u;
typedef __attribute__((ext_vector_type(8))) int          tdm8i;
typedef __attribute__((ext_vector_type(4))) int          tdm4i;

// 2D f32 tile load: tile_w elems x tile_h rows from a row-major tensor with
// given element stride, into LDS at byte offset lds_off.  D# per CDNA5 ISA.
__device__ __forceinline__ void tdm_load_tile_f32(uint32_t lds_off, const float* gptr,
                                                  int tile_w, int tile_h, int stride_elems) {
  uint64_t ga = (uint64_t)(uintptr_t)gptr;
  tdm4u g0;
  g0[0] = 1u;                                                    // count=1, user descriptor
  g0[1] = lds_off;                                               // lds_addr (bytes)
  g0[2] = (uint32_t)ga;                                          // global_addr[31:0]
  g0[3] = (uint32_t)((ga >> 32) & 0x1FFFFFFull) | (2u << 30);    // addr[56:32] | type=2
  tdm8i g1;
  g1[0] = (int)(2u << 16);                                       // data_size = 4B
  g1[1] = (int)(((uint32_t)stride_elems & 0xFFFFu) << 16);       // tensor_dim0[15:0]
  g1[2] = (int)((((uint32_t)stride_elems >> 16) & 0xFFFFu)
                | (((uint32_t)N_ & 0xFFFFu) << 16));             // td0 hi | tensor_dim1 lo
  g1[3] = (int)((((uint32_t)N_ >> 16) & 0xFFFFu)
                | ((uint32_t)tile_w << 16));                     // td1 hi | tile_dim0
  g1[4] = (int)((uint32_t)tile_h & 0xFFFFu);                     // tile_dim1 | tile_dim2=0
  g1[5] = stride_elems;                                          // tensor_dim0_stride lo32
  g1[6] = 0;
  g1[7] = 0;
  tdm4i g2 = {0, 0, 0, 0};
  tdm4i g3 = {0, 0, 0, 0};
  tdm8i gx = {0, 0, 0, 0, 0, 0, 0, 0};   // extra group in the 6-arg builtin form
  __builtin_amdgcn_tensor_load_to_lds(g0, g1, g2, g3, gx, 0);
}
#endif

// -------------------------------- 16-lane (DPP-row) reductions, reduce-to-all.
// xor1 = quad_perm(1,0,3,2)=0xB1, xor2 = quad_perm(2,3,0,1)=0x4E,
// cross-quad = row_half_mirror (0x141), cross-oct = row_mirror (0x140).
// DPP rows are 16 lanes, so the two half-waves reduce independently.
#if __has_builtin(__builtin_amdgcn_update_dpp)
template <int CTRL>
__device__ __forceinline__ float dpp_mv(float x) {
  return __int_as_float(
      __builtin_amdgcn_update_dpp(0, __float_as_int(x), CTRL, 0xF, 0xF, true));
}
__device__ __forceinline__ float rmax16(float v) {
  v = fmaxf(v, dpp_mv<0xB1>(v));
  v = fmaxf(v, dpp_mv<0x4E>(v));
  v = fmaxf(v, dpp_mv<0x141>(v));
  v = fmaxf(v, dpp_mv<0x140>(v));
  return v;
}
__device__ __forceinline__ float rsum16(float v) {
  v += dpp_mv<0xB1>(v);
  v += dpp_mv<0x4E>(v);
  v += dpp_mv<0x141>(v);
  v += dpp_mv<0x140>(v);
  return v;
}
#else
__device__ __forceinline__ float rmax16(float v) {
  v = fmaxf(v, __shfl_xor(v, 1, 32));
  v = fmaxf(v, __shfl_xor(v, 2, 32));
  v = fmaxf(v, __shfl_xor(v, 4, 32));
  v = fmaxf(v, __shfl_xor(v, 8, 32));
  return v;
}
__device__ __forceinline__ float rsum16(float v) {
  v += __shfl_xor(v, 1, 32);
  v += __shfl_xor(v, 2, 32);
  v += __shfl_xor(v, 4, 32);
  v += __shfl_xor(v, 8, 32);
  return v;
}
#endif

// out[b,n] = o_b (heads accumulate via atomics afterwards)
__global__ __launch_bounds__(256) void mha_init_out(float* out, const float* o_b) {
  int i = blockIdx.x * 256 + threadIdx.x;
  if (i < B_ * N_) out[i] = o_b[0];
}

// Batch-independent fold of o_w into the V weights:
//   c1[h,k] = w_v[k,:].ow_h   c2[h,k] = b_v[k,:].ow_h
__global__ __launch_bounds__(256) void mha_fold_ow(const float* __restrict__ qkv_w,
                                                   const float* __restrict__ qkv_b,
                                                   const float* __restrict__ o_w,
                                                   float* __restrict__ c1,
                                                   float* __restrict__ c2) {
  int idx = blockIdx.x * 256 + threadIdx.x;   // H*N threads
  int h = idx >> 11, k = idx & (N_ - 1);
  const float* wr = qkv_w + (size_t)k * C3E + h * 192 + 128;   // V block of head h
  const float* br = qkv_b + (size_t)k * C3E + h * 192 + 128;
  const float* ow = o_w + h * HD_;
  float a = 0.f, bb = 0.f;
  #pragma unroll 8
  for (int d = 0; d < HD_; ++d) {
    a  = fmaf(wr[d], ow[d], a);
    bb = fmaf(br[d], ow[d], bb);
  }
  c1[idx] = a;
  c2[idx] = bb;
}

// Flash attention with the PV GEMM collapsed to scalar weights s[k].
__global__ __launch_bounds__(256) void mha_attn(const float* __restrict__ x,
                                                const float* __restrict__ qkv_w,
                                                const float* __restrict__ qkv_b,
                                                const float* __restrict__ c1,
                                                const float* __restrict__ c2,
                                                float* __restrict__ out) {
  __shared__ __align__(16) float  wst[CHUNK * HD_];   // staged w K-slice (f32)
  __shared__ __align__(16) float  bst[CHUNK * HD_];   // staged b K-slice (f32)
  __shared__ __align__(16) __bf16 Kbf[CHUNK * HD_];   // K chunk in bf16
  __shared__ float xs[CHUNK];
  __shared__ float sv[CHUNK];                         // s[k] = V[k,:].ow_h

  const int tid  = threadIdx.x;
  const int lane = tid & 31;
  const int wave = tid >> 5;
  const int hf   = lane >> 4;     // half-wave
  const int mrow = lane & 15;     // row (A) / column (B) index

  const int b     = blockIdx.x >> 4;
  const int h     = blockIdx.x & 15;
  const int bN    = b * N_;
  const int hN    = h * N_;
  const int hbase = h * 192;

  // ---- Build resident Q A-fragments (scale 1/8 folded in).
  const int   qRow = blockIdx.y * QT + wave * 16 + mrow;
  const float xq   = x[bN + qRow] * 0.125f;
  const float* wrow = qkv_w + (size_t)qRow * C3E + hbase;
  const float* brow = qkv_b + (size_t)qRow * C3E + hbase;
  v16bf aF0, aF1;
  #pragma unroll
  for (int e = 0; e < 8; ++e) {
    int d1 = hf * 8 + e;
    aF0[e]     = (__bf16)fmaf(xq, wrow[d1],      0.125f * brow[d1]);
    aF0[e + 8] = (__bf16)fmaf(xq, wrow[d1 + 16], 0.125f * brow[d1 + 16]);
    aF1[e]     = (__bf16)fmaf(xq, wrow[d1 + 32], 0.125f * brow[d1 + 32]);
    aF1[e + 8] = (__bf16)fmaf(xq, wrow[d1 + 48], 0.125f * brow[d1 + 48]);
  }

  // Online softmax state: lane holds rows hf*8 + r (matches C-fragment rows).
  float mR[8], lR[8], oR[8];
  #pragma unroll
  for (int r = 0; r < 8; ++r) { mR[r] = -1e30f; lR[r] = 0.f; oR[r] = 0.f; }

  for (int k0 = 0; k0 < N_; k0 += CHUNK) {
    // ---- Phase A: stage the f32 K-slices (cols hbase+64..+127) via TDM.
#if USE_TDM
    if (wave == 0) {
      tdm_load_tile_f32((uint32_t)(uintptr_t)wst,
                        qkv_w + (size_t)k0 * C3E + hbase + 64, HD_, CHUNK, C3E);
      tdm_load_tile_f32((uint32_t)(uintptr_t)bst,
                        qkv_b + (size_t)k0 * C3E + hbase + 64, HD_, CHUNK, C3E);
      __builtin_amdgcn_s_wait_tensorcnt(0);
    }
#else
    {
      const float* gw = qkv_w + (size_t)k0 * C3E + hbase + 64;
      const float* gb = qkv_b + (size_t)k0 * C3E + hbase + 64;
      for (int i = tid; i < CHUNK * HD_; i += 256) {
        int r = i >> 6, d = i & 63;
        wst[i] = gw[(size_t)r * C3E + d];
        bst[i] = gb[(size_t)r * C3E + d];
      }
    }
#endif
    if (tid >= 32 && tid < 32 + CHUNK) xs[tid - 32] = x[bN + k0 + (tid - 32)];
    __syncthreads();

    // ---- Phase B: K -> bf16, and s[k] from the folded V dot products.
    for (int i = tid; i < CHUNK * HD_; i += 256) {
      int r = i >> 6;
      Kbf[i] = (__bf16)fmaf(xs[r], wst[i], bst[i]);
    }
    if (tid < CHUNK)
      sv[tid] = fmaf(xs[tid], c1[hN + k0 + tid], c2[hN + k0 + tid]);
    __syncthreads();

    // ---- Phase C: 4 independent WMMA tile-pairs, then ONE softmax pass/chunk.
    v8f   cf[4];
    float sv4[4];
    #pragma unroll
    for (int kt = 0; kt < CHUNK / 16; ++kt) {
      union { uint4 u[2]; v16bf v; } b0, b1;
      const __bf16* kr = &Kbf[(kt * 16 + mrow) * HD_ + hf * 8];
      b0.u[0] = *(const uint4*)(kr);        // d =  hf*8 .. +8
      b0.u[1] = *(const uint4*)(kr + 16);   // d = 16+hf*8
      b1.u[0] = *(const uint4*)(kr + 32);
      b1.u[1] = *(const uint4*)(kr + 48);

      v8f c;
      #pragma unroll
      for (int r = 0; r < 8; ++r) c[r] = 0.f;
      c = __builtin_amdgcn_wmma_f32_16x16x32_bf16(false, aF0, false, b0.v,
                                                  (short)0, c, false, false);
      c = __builtin_amdgcn_wmma_f32_16x16x32_bf16(false, aF1, false, b1.v,
                                                  (short)0, c, false, false);
      cf[kt]  = c;
      sv4[kt] = sv[kt * 16 + mrow];
    }

    #pragma unroll
    for (int r = 0; r < 8; ++r) {
      float l0 = cf[0][r], l1 = cf[1][r], l2 = cf[2][r], l3 = cf[3][r];
      float lm = fmaxf(fmaxf(l0, l1), fmaxf(l2, l3));
      float mn = fmaxf(mR[r], rmax16(lm));
      float p0 = __expf(l0 - mn);
      float p1 = __expf(l1 - mn);
      float p2 = __expf(l2 - mn);
      float p3 = __expf(l3 - mn);
      float psl = (p0 + p1) + (p2 + p3);
      float pwl = fmaf(p0, sv4[0], fmaf(p1, sv4[1], fmaf(p2, sv4[2], p3 * sv4[3])));
      float ps = rsum16(psl);
      float pw = rsum16(pwl);
      float corr = __expf(mR[r] - mn);
      lR[r] = fmaf(lR[r], corr, ps);
      oR[r] = fmaf(oR[r], corr, pw);
      mR[r] = mn;
    }
    __syncthreads();
  }

  // ---- Epilogue: one lane per half writes its 8 rows; heads accumulate.
  if (mrow == 0) {
    int rowBase = bN + blockIdx.y * QT + wave * 16 + hf * 8;
    #pragma unroll
    for (int r = 0; r < 8; ++r)
      atomicAdd(&out[rowBase + r], oR[r] / lR[r]);
  }
}

extern "C" void kernel_launch(void* const* d_in, const int* in_sizes, int n_in,
                              void* d_out, int out_size, void* d_ws, size_t ws_size,
                              hipStream_t stream) {
  (void)in_sizes; (void)n_in; (void)out_size; (void)ws_size;
  const float* x     = (const float*)d_in[0];
  const float* qkv_w = (const float*)d_in[1];
  const float* qkv_b = (const float*)d_in[2];
  const float* o_w   = (const float*)d_in[3];
  const float* o_b   = (const float*)d_in[4];
  float* out = (float*)d_out;
  float* c1  = (float*)d_ws;            // [H, N]
  float* c2  = c1 + H_ * N_;            // [H, N]  (256 KB total workspace)

  mha_init_out<<<(B_ * N_ + 255) / 256, 256, 0, stream>>>(out, o_b);
  mha_fold_ow<<<(H_ * N_) / 256, 256, 0, stream>>>(qkv_w, qkv_b, o_w, c1, c2);
  mha_attn<<<dim3(B_ * H_, N_ / QT), 256, 0, stream>>>(x, qkv_w, qkv_b, c1, c2, out);
}
